// MultiHeadAttention_78116865180269
// MI455X (gfx1250) — matrix operand order repaired and checked by reading, not compiled, not run on hardware
//
#include <hip/hip_runtime.h>
#include <hip/hip_bf16.h>

// ---------------------------------------------------------------------------
// MI455X (gfx1250) multi-head attention, bf16 WMMA pipeline.
//   B=2, SQ=SK=2048, QDIM=KDIM=HID=1024, H=16, DH=64
// Stages: pack weights (bf16, transposed) -> per-head projections (WMMA,
//         32x64 tiles/wave) -> flash attention (WMMA + online softmax)
//         -> output projection (WMMA, 32x64 tiles/wave).
// All matrix math uses v_wmma_f32_16x16x32_bf16 (wave32).
// fp32->bf16 packing via v_perm_b32 (1 op) + round-half-up bias.
// ---------------------------------------------------------------------------

typedef __attribute__((ext_vector_type(16))) __bf16 v16bf;
typedef __attribute__((ext_vector_type(8)))  float  v8f;

#define WMMA_BF16(A, B, C) \
  __builtin_amdgcn_wmma_f32_16x16x32_bf16(false, (A), false, (B), (short)0, (C), false, false)

union Frag {
  unsigned int u[8];
  uint4        q[2];
  v16bf        v;
};

// --- fp32 -> bf16: v_perm_b32 pack (round-half-up), d16_hi-foldable scalar ---
__device__ __forceinline__ unsigned int packbf2(float a, float b) {
#if __has_builtin(__builtin_amdgcn_perm)
  unsigned int ua = __float_as_uint(a) + 0x8000u;
  unsigned int ub = __float_as_uint(b) + 0x8000u;
  // result bytes: [1:0] = ua[3:2] (sel 3,2), [3:2] = ub[3:2] (sel 7,6)
  return __builtin_amdgcn_perm(ub, ua, 0x07060302u);
#else
  unsigned int x = __float_as_uint(a), y = __float_as_uint(b);
  x += 0x7FFFu + ((x >> 16) & 1u);
  y += 0x7FFFu + ((y >> 16) & 1u);
  return (x >> 16) | (y & 0xFFFF0000u);
#endif
}
__device__ __forceinline__ unsigned short f2bf(float f) {
  // backend folds ">>16" 16-bit store into global/ds store_d16_hi_b16
  return (unsigned short)((__float_as_uint(f) + 0x8000u) >> 16);
}

// dims
#define NB     2
#define NH     16
#define SEQ    2048
#define DMODEL 1024
#define DH     64

// ---------------------------------------------------------------------------
// Kernel 1: weights fp32 -> bf16, transposed so WMMA B-operand K-pairs are
// contiguous.  WqT/WkT/WvT: [H][DH=64][1024],  WoT: [1024(n)][1024(k)].
// ---------------------------------------------------------------------------
__global__ void mha_pack_weights(const float* __restrict__ Wq,
                                 const float* __restrict__ Wk,
                                 const float* __restrict__ Wv,
                                 const float* __restrict__ Wo,
                                 unsigned short* __restrict__ wqT,
                                 unsigned short* __restrict__ wkT,
                                 unsigned short* __restrict__ wvT,
                                 unsigned short* __restrict__ woT) {
  int i = blockIdx.x * blockDim.x + threadIdx.x;
  const int NW = NH * DMODEL * DH;               // per projection matrix
  if (i < 3 * NW) {
    int m = i / NW, r = i % NW;
    int h = r / (DMODEL * DH);
    int rr = r % (DMODEL * DH);
    int d = rr / DH, e = rr % DH;
    const float* src = (m == 0) ? Wq : ((m == 1) ? Wk : Wv);
    unsigned short* dst = (m == 0) ? wqT : ((m == 1) ? wkT : wvT);
    dst[((size_t)h * DH + e) * DMODEL + d] = f2bf(src[((size_t)h * DMODEL + d) * DH + e]);
  } else {
    int r = i - 3 * NW;                          // 0 .. 1024*1024-1
    int d = r >> 10, n = r & 1023;
    woT[(size_t)n * DMODEL + d] = f2bf(Wo[(size_t)d * DMODEL + n]);
  }
}

// ---------------------------------------------------------------------------
// Kernel 2: per-head projections.  One wave = one 32x64 output tile (two
// 16-row A-frags share every weight B-frag -> 2x WMMA per B load).
//   qh, kh : [B*H][SEQ][64] bf16
//   vhT    : [B*H][64][SEQ] bf16   (transposed: B-operand for P*V)
// ---------------------------------------------------------------------------
__global__ void __launch_bounds__(256) mha_project(
    const float* __restrict__ q, const float* __restrict__ k, const float* __restrict__ v,
    const unsigned short* __restrict__ wqT, const unsigned short* __restrict__ wkT,
    const unsigned short* __restrict__ wvT,
    unsigned short* __restrict__ qh, unsigned short* __restrict__ kh,
    unsigned short* __restrict__ vhT) {
  int lane  = threadIdx.x & 31;
  int gwave = blockIdx.x * (blockDim.x >> 5) + (threadIdx.x >> 5);
  int mt  = gwave & 63;             // 32-row tile within SEQ
  int h   = (gwave >> 6) & 15;
  int b   = (gwave >> 10) & 1;
  int mat = gwave >> 11;            // 0=q 1=k 2=v
  int hl = lane >> 4, ln = lane & 15;
  int row0 = mt * 32;

  const float* X = (mat == 0) ? q : ((mat == 1) ? k : v);
  const unsigned short* WT =
      ((mat == 0) ? wqT : ((mat == 1) ? wkT : wvT)) + (size_t)h * DH * DMODEL;

  v8f c00 = {}, c01 = {}, c02 = {}, c03 = {};     // rows row0..row0+15
  v8f c10 = {}, c11 = {}, c12 = {}, c13 = {};     // rows row0+16..row0+31
  const float* xrow0 = X + ((size_t)(b * SEQ + row0 + ln)) * DMODEL + hl * 16;
  const float* xrow1 = xrow0 + (size_t)16 * DMODEL;
  const unsigned short* wrow = WT + (size_t)ln * DMODEL + hl * 16;

  for (int kb = 0; kb < DMODEL; kb += 32) {
    Frag a0, a1;
    {
      float4 x0 = *(const float4*)(xrow0 + kb);
      float4 x1 = *(const float4*)(xrow0 + kb + 4);
      float4 x2 = *(const float4*)(xrow0 + kb + 8);
      float4 x3 = *(const float4*)(xrow0 + kb + 12);
      a0.u[0] = packbf2(x0.x, x0.y); a0.u[1] = packbf2(x0.z, x0.w);
      a0.u[2] = packbf2(x1.x, x1.y); a0.u[3] = packbf2(x1.z, x1.w);
      a0.u[4] = packbf2(x2.x, x2.y); a0.u[5] = packbf2(x2.z, x2.w);
      a0.u[6] = packbf2(x3.x, x3.y); a0.u[7] = packbf2(x3.z, x3.w);
    }
    {
      float4 x0 = *(const float4*)(xrow1 + kb);
      float4 x1 = *(const float4*)(xrow1 + kb + 4);
      float4 x2 = *(const float4*)(xrow1 + kb + 8);
      float4 x3 = *(const float4*)(xrow1 + kb + 12);
      a1.u[0] = packbf2(x0.x, x0.y); a1.u[1] = packbf2(x0.z, x0.w);
      a1.u[2] = packbf2(x1.x, x1.y); a1.u[3] = packbf2(x1.z, x1.w);
      a1.u[4] = packbf2(x2.x, x2.y); a1.u[5] = packbf2(x2.z, x2.w);
      a1.u[6] = packbf2(x3.x, x3.y); a1.u[7] = packbf2(x3.z, x3.w);
    }
    Frag b0, b1, b2, b3;
    const unsigned short* wb = wrow + kb;
    b0.q[0] = *(const uint4*)(wb);                 b0.q[1] = *(const uint4*)(wb + 8);
    b1.q[0] = *(const uint4*)(wb + 16 * DMODEL);   b1.q[1] = *(const uint4*)(wb + 16 * DMODEL + 8);
    b2.q[0] = *(const uint4*)(wb + 32 * DMODEL);   b2.q[1] = *(const uint4*)(wb + 32 * DMODEL + 8);
    b3.q[0] = *(const uint4*)(wb + 48 * DMODEL);   b3.q[1] = *(const uint4*)(wb + 48 * DMODEL + 8);

    c00 = WMMA_BF16(a0.v, b0.v, c00);  c10 = WMMA_BF16(a1.v, b0.v, c10);
    c01 = WMMA_BF16(a0.v, b1.v, c01);  c11 = WMMA_BF16(a1.v, b1.v, c11);
    c02 = WMMA_BF16(a0.v, b2.v, c02);  c12 = WMMA_BF16(a1.v, b2.v, c12);
    c03 = WMMA_BF16(a0.v, b3.v, c03);  c13 = WMMA_BF16(a1.v, b3.v, c13);
  }

  if (mat < 2) {
    unsigned short* dst = ((mat == 0) ? qh : kh) + ((size_t)(b * NH + h)) * SEQ * DH;
#pragma unroll
    for (int r = 0; r < 8; r++) {
      int ra = row0 + r + hl * 8;
      int rb = ra + 16;
      dst[(size_t)ra * DH + ln]      = f2bf(c00[r]);
      dst[(size_t)ra * DH + 16 + ln] = f2bf(c01[r]);
      dst[(size_t)ra * DH + 32 + ln] = f2bf(c02[r]);
      dst[(size_t)ra * DH + 48 + ln] = f2bf(c03[r]);
      dst[(size_t)rb * DH + ln]      = f2bf(c10[r]);
      dst[(size_t)rb * DH + 16 + ln] = f2bf(c11[r]);
      dst[(size_t)rb * DH + 32 + ln] = f2bf(c12[r]);
      dst[(size_t)rb * DH + 48 + ln] = f2bf(c13[r]);
    }
  } else {
    unsigned short* dst = vhT + ((size_t)(b * NH + h)) * DH * SEQ;   // [d][s]
#pragma unroll
    for (int r = 0; r < 8; r++) {
      int ra = row0 + r + hl * 8;
      int rb = ra + 16;
      dst[(size_t)(ln) * SEQ + ra]      = f2bf(c00[r]);
      dst[(size_t)(16 + ln) * SEQ + ra] = f2bf(c01[r]);
      dst[(size_t)(32 + ln) * SEQ + ra] = f2bf(c02[r]);
      dst[(size_t)(48 + ln) * SEQ + ra] = f2bf(c03[r]);
      dst[(size_t)(ln) * SEQ + rb]      = f2bf(c10[r]);
      dst[(size_t)(16 + ln) * SEQ + rb] = f2bf(c11[r]);
      dst[(size_t)(32 + ln) * SEQ + rb] = f2bf(c12[r]);
      dst[(size_t)(48 + ln) * SEQ + rb] = f2bf(c13[r]);
    }
  }
}

// ---------------------------------------------------------------------------
// Kernel 3: flash attention.  One wave = one (b,h,16-row q-tile).
// Chunks of 64 keys: 8 WMMAs scores, scale+mask, online softmax with
// shfl_xor row reductions, P -> LDS (C-layout -> A-layout), 8 WMMAs P*V.
// Prefetch next K/V chunk to overlap the softmax VALU phase.
// ---------------------------------------------------------------------------
__global__ void __launch_bounds__(256) mha_attn(
    const unsigned short* __restrict__ qh, const unsigned short* __restrict__ kh,
    const unsigned short* __restrict__ vhT, const unsigned char* __restrict__ mask,
    unsigned short* __restrict__ att) {
  __shared__ unsigned short pbuf[8 * 16 * DH];   // 2KB per wave
  int lane = threadIdx.x & 31;
  int wave = threadIdx.x >> 5;
  int gwave = blockIdx.x * 8 + wave;
  int qt = gwave & 127;
  int h  = (gwave >> 7) & 15;
  int b  = gwave >> 11;
  int hl = lane >> 4, ln = lane & 15;
  int row0 = qt * 16;

  const unsigned short* Qh = qh + ((size_t)(b * NH + h)) * SEQ * DH;
  const unsigned short* Kh = kh + ((size_t)(b * NH + h)) * SEQ * DH;
  const unsigned short* Vt = vhT + ((size_t)(b * NH + h)) * DH * SEQ;
  const unsigned char*  M  = mask + ((size_t)b * SEQ + row0) * SEQ;
  unsigned short* pb = pbuf + wave * 16 * DH;

  // q-tile A-frags for both K-steps (DH=64 -> two 16x32 slabs), loaded once.
  Frag aq0, aq1;
  {
    const unsigned short* qr = Qh + (size_t)(row0 + ln) * DH + hl * 16;
    aq0.q[0] = *(const uint4*)(qr);      aq0.q[1] = *(const uint4*)(qr + 8);
    aq1.q[0] = *(const uint4*)(qr + 32); aq1.q[1] = *(const uint4*)(qr + 40);
  }

  float mrow[8], lrow[8];
#pragma unroll
  for (int r = 0; r < 8; r++) { mrow[r] = -3.0e38f; lrow[r] = 0.f; }
  v8f acc0 = {}, acc1 = {}, acc2 = {}, acc3 = {};

  for (int kb = 0; kb < SEQ; kb += 64) {
    // prefetch next chunk of K and V (speculative; dropped if OOB)
    __builtin_prefetch(Kh + (size_t)(kb + 64 + ln) * DH, 0, 1);
    __builtin_prefetch(Vt + (size_t)ln * SEQ + kb + 64, 0, 1);

    v8f s0 = {}, s1 = {}, s2 = {}, s3 = {};
    // ---- scores: S = Q * K^T ----
#pragma unroll
    for (int s = 0; s < 2; s++) {
      const unsigned short* kbase = Kh + (size_t)(kb + ln) * DH + s * 32 + hl * 16;
      Frag b0, b1, b2, b3;
      b0.q[0] = *(const uint4*)(kbase);            b0.q[1] = *(const uint4*)(kbase + 8);
      b1.q[0] = *(const uint4*)(kbase + 16 * DH);  b1.q[1] = *(const uint4*)(kbase + 16 * DH + 8);
      b2.q[0] = *(const uint4*)(kbase + 32 * DH);  b2.q[1] = *(const uint4*)(kbase + 32 * DH + 8);
      b3.q[0] = *(const uint4*)(kbase + 48 * DH);  b3.q[1] = *(const uint4*)(kbase + 48 * DH + 8);
      const v16bf A = s ? aq1.v : aq0.v;
      s0 = WMMA_BF16(A, b0.v, s0);
      s1 = WMMA_BF16(A, b1.v, s1);
      s2 = WMMA_BF16(A, b2.v, s2);
      s3 = WMMA_BF16(A, b3.v, s3);
    }
    // ---- scale (1/sqrt(64)) + mask ----
#pragma unroll
    for (int r = 0; r < 8; r++) {
      int qr = r + hl * 8;
      const unsigned char* mr = M + (size_t)qr * SEQ + kb + ln;
      float v0 = s0[r] * 0.125f; if (mr[0])  v0 = -3.0e38f; s0[r] = v0;
      float v1 = s1[r] * 0.125f; if (mr[16]) v1 = -3.0e38f; s1[r] = v1;
      float v2 = s2[r] * 0.125f; if (mr[32]) v2 = -3.0e38f; s2[r] = v2;
      float v3 = s3[r] * 0.125f; if (mr[48]) v3 = -3.0e38f; s3[r] = v3;
    }
    // ---- online softmax (rows live in 16-lane halves) ----
#pragma unroll
    for (int r = 0; r < 8; r++) {
      float mx = fmaxf(fmaxf(s0[r], s1[r]), fmaxf(s2[r], s3[r]));
      mx = fmaxf(mx, __shfl_xor(mx, 1, 16));
      mx = fmaxf(mx, __shfl_xor(mx, 2, 16));
      mx = fmaxf(mx, __shfl_xor(mx, 4, 16));
      mx = fmaxf(mx, __shfl_xor(mx, 8, 16));
      float mn = fmaxf(mrow[r], mx);
      float alpha = __expf(mrow[r] - mn);
      mrow[r] = mn;
      float p0 = __expf(s0[r] - mn), p1 = __expf(s1[r] - mn);
      float p2 = __expf(s2[r] - mn), p3 = __expf(s3[r] - mn);
      float ls = p0 + p1 + p2 + p3;
      ls += __shfl_xor(ls, 1, 16);
      ls += __shfl_xor(ls, 2, 16);
      ls += __shfl_xor(ls, 4, 16);
      ls += __shfl_xor(ls, 8, 16);
      lrow[r] = lrow[r] * alpha + ls;
      acc0[r] *= alpha; acc1[r] *= alpha; acc2[r] *= alpha; acc3[r] *= alpha;
      int qr = r + hl * 8;
      pb[qr * DH + ln]      = f2bf(p0);
      pb[qr * DH + 16 + ln] = f2bf(p1);
      pb[qr * DH + 32 + ln] = f2bf(p2);
      pb[qr * DH + 48 + ln] = f2bf(p3);
    }
    asm volatile("s_wait_dscnt 0" ::: "memory");   // cross-lane LDS RAW
    // ---- O += P * V ----
#pragma unroll
    for (int s = 0; s < 2; s++) {
      Frag ap;
      const unsigned short* prow = pb + ln * DH + s * 32 + hl * 16;
      ap.q[0] = *(const uint4*)(prow);
      ap.q[1] = *(const uint4*)(prow + 8);
      const unsigned short* vb = Vt + (size_t)ln * SEQ + kb + s * 32 + hl * 16;
      Frag b0, b1, b2, b3;
      b0.q[0] = *(const uint4*)(vb);             b0.q[1] = *(const uint4*)(vb + 8);
      b1.q[0] = *(const uint4*)(vb + 16 * SEQ);  b1.q[1] = *(const uint4*)(vb + 16 * SEQ + 8);
      b2.q[0] = *(const uint4*)(vb + 32 * SEQ);  b2.q[1] = *(const uint4*)(vb + 32 * SEQ + 8);
      b3.q[0] = *(const uint4*)(vb + 48 * SEQ);  b3.q[1] = *(const uint4*)(vb + 48 * SEQ + 8);
      acc0 = WMMA_BF16(ap.v, b0.v, acc0);
      acc1 = WMMA_BF16(ap.v, b1.v, acc1);
      acc2 = WMMA_BF16(ap.v, b2.v, acc2);
      acc3 = WMMA_BF16(ap.v, b3.v, acc3);
    }
  }

  // epilogue: normalize and store head-concat bf16
  unsigned short* arow = att + ((size_t)(b * SEQ + row0)) * DMODEL + h * DH;
#pragma unroll
  for (int r = 0; r < 8; r++) {
    float inv = 1.0f / lrow[r];
    int qr = r + hl * 8;
    arow[(size_t)qr * DMODEL + ln]      = f2bf(acc0[r] * inv);
    arow[(size_t)qr * DMODEL + 16 + ln] = f2bf(acc1[r] * inv);
    arow[(size_t)qr * DMODEL + 32 + ln] = f2bf(acc2[r] * inv);
    arow[(size_t)qr * DMODEL + 48 + ln] = f2bf(acc3[r] * inv);
  }
}

// ---------------------------------------------------------------------------
// Kernel 4: output projection  out = att(bf16) @ Wo  -> fp32 d_out.
// One wave = 32x64 tile (B-frags reused across two row tiles).
// ---------------------------------------------------------------------------
__global__ void __launch_bounds__(256) mha_outproj(
    const unsigned short* __restrict__ att, const unsigned short* __restrict__ woT,
    float* __restrict__ out) {
  int lane  = threadIdx.x & 31;
  int gwave = blockIdx.x * (blockDim.x >> 5) + (threadIdx.x >> 5);
  int nt = gwave & 15;         // 64-col tile
  int mt = gwave >> 4;         // 32-row tile (0..127)
  int hl = lane >> 4, ln = lane & 15;
  int row0 = mt * 32, n0 = nt * 64;

  v8f c00 = {}, c01 = {}, c02 = {}, c03 = {};
  v8f c10 = {}, c11 = {}, c12 = {}, c13 = {};
  const unsigned short* arow0 = att + (size_t)(row0 + ln) * DMODEL + hl * 16;
  const unsigned short* arow1 = arow0 + (size_t)16 * DMODEL;
  const unsigned short* wrow  = woT + (size_t)(n0 + ln) * DMODEL + hl * 16;

  for (int kb = 0; kb < DMODEL; kb += 32) {
    Frag a0, a1;
    a0.q[0] = *(const uint4*)(arow0 + kb);
    a0.q[1] = *(const uint4*)(arow0 + kb + 8);
    a1.q[0] = *(const uint4*)(arow1 + kb);
    a1.q[1] = *(const uint4*)(arow1 + kb + 8);
    const unsigned short* wb = wrow + kb;
    Frag b0, b1, b2, b3;
    b0.q[0] = *(const uint4*)(wb);                 b0.q[1] = *(const uint4*)(wb + 8);
    b1.q[0] = *(const uint4*)(wb + 16 * DMODEL);   b1.q[1] = *(const uint4*)(wb + 16 * DMODEL + 8);
    b2.q[0] = *(const uint4*)(wb + 32 * DMODEL);   b2.q[1] = *(const uint4*)(wb + 32 * DMODEL + 8);
    b3.q[0] = *(const uint4*)(wb + 48 * DMODEL);   b3.q[1] = *(const uint4*)(wb + 48 * DMODEL + 8);
    c00 = WMMA_BF16(a0.v, b0.v, c00);  c10 = WMMA_BF16(a1.v, b0.v, c10);
    c01 = WMMA_BF16(a0.v, b1.v, c01);  c11 = WMMA_BF16(a1.v, b1.v, c11);
    c02 = WMMA_BF16(a0.v, b2.v, c02);  c12 = WMMA_BF16(a1.v, b2.v, c12);
    c03 = WMMA_BF16(a0.v, b3.v, c03);  c13 = WMMA_BF16(a1.v, b3.v, c13);
  }
#pragma unroll
  for (int r = 0; r < 8; r++) {
    int ra = row0 + r + hl * 8;
    int rb = ra + 16;
    out[(size_t)ra * DMODEL + n0 + ln]      = c00[r];
    out[(size_t)ra * DMODEL + n0 + 16 + ln] = c01[r];
    out[(size_t)ra * DMODEL + n0 + 32 + ln] = c02[r];
    out[(size_t)ra * DMODEL + n0 + 48 + ln] = c03[r];
    out[(size_t)rb * DMODEL + n0 + ln]      = c10[r];
    out[(size_t)rb * DMODEL + n0 + 16 + ln] = c11[r];
    out[(size_t)rb * DMODEL + n0 + 32 + ln] = c12[r];
    out[(size_t)rb * DMODEL + n0 + 48 + ln] = c13[r];
  }
}

// ---------------------------------------------------------------------------
extern "C" void kernel_launch(void* const* d_in, const int* in_sizes, int n_in,
                              void* d_out, int out_size, void* d_ws, size_t ws_size,
                              hipStream_t stream) {
  const float*         q    = (const float*)d_in[0];
  const float*         k    = (const float*)d_in[1];
  const float*         v    = (const float*)d_in[2];
  const unsigned char* mask = (const unsigned char*)d_in[3];
  const float*         Wq   = (const float*)d_in[4];
  const float*         Wk   = (const float*)d_in[5];
  const float*         Wv   = (const float*)d_in[6];
  const float*         Wo   = (const float*)d_in[7];
  float* out = (float*)d_out;

  // workspace layout (bf16 elements), total ~40 MB
  unsigned short* ws  = (unsigned short*)d_ws;
  unsigned short* wqT = ws;                       // 1M elems each
  unsigned short* wkT = wqT + (1u << 20);
  unsigned short* wvT = wkT + (1u << 20);
  unsigned short* woT = wvT + (1u << 20);
  unsigned short* qh  = woT + (1u << 20);         // 4M elems each
  unsigned short* kh  = qh  + (1u << 22);
  unsigned short* vhT = kh  + (1u << 22);
  unsigned short* att = vhT + (1u << 22);

  mha_pack_weights<<<16384, 256, 0, stream>>>(Wq, Wk, Wv, Wo, wqT, wkT, wvT, woT);
  mha_project<<<768, 256, 0, stream>>>(q, k, v, wqT, wkT, wvT, qh, kh, vhT);
  mha_attn<<<512, 256, 0, stream>>>(qh, kh, vhT, mask, att);
  mha_outproj<<<256, 256, 0, stream>>>(att, woT, out);
}
